// ScanModuleV8_5978594476776
// MI455X (gfx1250) — matrix operand order repaired and checked by reading, not compiled, not run on hardware
//
#include <hip/hip_runtime.h>
#include <hip/hip_bf16.h>

typedef __attribute__((ext_vector_type(16))) __bf16 v16bf;
typedef __attribute__((ext_vector_type(8)))  __bf16 v8bf;
typedef __attribute__((ext_vector_type(8)))  float  v8f;
typedef __attribute__((ext_vector_type(4)))  unsigned int v4u;
typedef __attribute__((ext_vector_type(8)))  int v8i_;
typedef __attribute__((ext_vector_type(4)))  int v4i_;

#define WMMA_BF16(a, b, c) \
  __builtin_amdgcn_wmma_f32_16x16x32_bf16(false, (a), false, (b), (short)0, (c), false, false)

#if defined(__has_builtin)
#  if __has_builtin(__builtin_amdgcn_tensor_load_to_lds) && \
      __has_builtin(__builtin_amdgcn_s_wait_tensorcnt)
#    define HAVE_TDM 1
#  endif
#  if __has_builtin(__builtin_amdgcn_tanhf)
#    define FAST_TANH(x) __builtin_amdgcn_tanhf(x)
#  endif
#endif
#ifndef FAST_TANH
#  define FAST_TANH(x) tanhf(x)
#endif

static constexpr int BB   = 64;    // batch
static constexpr int TT   = 1024;  // time
static constexpr int DD   = 256;   // feature dim
static constexpr int XPAD = 264;   // padded bf16 row stride for A-layout LDS (bank spread)

// ---- A fragment (16x32 bf16, row-major source in LDS with row stride `stride`) ----
// Layout: lanes 0-15 -> M=lane, K 0..15; lanes 16-31 -> M=lane-16,
// K 16..31. Each chunk = 8 contiguous bf16 = one b128 LDS load.
__device__ inline v16bf load_A_frag(const __bf16* lds, int row0, int k0, int stride) {
  const int lane = threadIdx.x & 31;
  const int l16  = lane & 15;
  const int hi   = (lane >> 4) & 1;
  const __bf16* p = lds + (row0 + l16) * stride + k0 + hi * 16;
  union { v16bf v; v8bf h[2]; } u;
  u.h[0] = *(const v8bf*)(p);
  u.h[1] = *(const v8bf*)(p + 8);
  return u.v;
}

// ---- B fragment (32x16 bf16) from LDS pre-swizzled into fragment order ----
__device__ inline v16bf load_B_frag(const __bf16* us, int ktile, int ntile) {
  const int lane = threadIdx.x & 31;
  const int l16  = lane & 15;
  const int hi   = (lane >> 4) & 1;
  const __bf16* p = us + ktile * (DD * 32) + (ntile * 16 + l16) * 32 + hi * 16;
  union { v16bf v; v8bf h[2]; } u;
  u.h[0] = *(const v8bf*)(p);
  u.h[1] = *(const v8bf*)(p + 8);
  return u.v;
}

#if defined(HAVE_TDM)
// Issue one TDM transfer: 16384 contiguous f32 (64 KB) global -> LDS.
// D# group0: count=1 | lds_addr | global_addr[56:0] | type=2 ("image").
// D# group1: data_size=4B; 2D tensor 16384x1, tile 16384x1; stride0=16384.
// clang-23 toolchain form: 6 args (g0, g1, g2, g3, g4, cpol); unused groups zero.
__device__ inline void tdm_load_slice(const float* gsrc, unsigned lds_off) {
  const unsigned long long ga = (unsigned long long)(uintptr_t)gsrc;
  v4u g0;
  g0[0] = 1u;                                            // count=1 (valid user D#)
  g0[1] = lds_off;                                       // lds_addr (bytes)
  g0[2] = (unsigned)(ga & 0xFFFFFFFFu);                  // global_addr[31:0]
  g0[3] = (unsigned)((ga >> 32) & 0x01FFFFFFu)           // global_addr[56:32]
        | (2u << 30);                                    // type=2
  v8i_ g1;
  g1[0] = (int)(2u << 16);                               // data_size=2 (4 bytes)
  g1[1] = (int)((16384u & 0xFFFFu) << 16);               // tensor_dim0[15:0]
  g1[2] = (int)((16384u >> 16) | (1u << 16));            // tensor_dim0[31:16] | tensor_dim1 lo=1
  g1[3] = (int)(16384u << 16);                           // tensor_dim1 hi=0 | tile_dim0=16384
  g1[4] = 1;                                             // tile_dim1=1, tile_dim2=0
  g1[5] = (int)16384u;                                   // tensor_dim0_stride[31:0]
  g1[6] = 0;                                             // stride0 hi | stride1 lo
  g1[7] = 0;                                             // stride1 hi
  const v4i_ z4 = {0, 0, 0, 0};
  const v8i_ z8 = {0, 0, 0, 0, 0, 0, 0, 0};
  __builtin_amdgcn_tensor_load_to_lds(g0, g1, z4, z4, z8, 0);
}
#endif

// =====================================================================
// Kernel 1: Xu[t,b,:] = x[b,t,:] @ U + bias   (parallel pre-pass GEMM)
// =====================================================================
__global__ __launch_bounds__(256) void xu_gemm_kernel(
    const float* __restrict__ x, const float* __restrict__ U,
    const float* __restrict__ bias, float* __restrict__ Xu) {
  __shared__ __bf16 xs[64 * XPAD];      // 64 rows of x in bf16,  ~34 KB
  __shared__ __bf16 us[4 * DD * 32];    // half of U in frag order, 64 KB

  const int tid = threadIdx.x;
  const int r0  = blockIdx.x * 64;      // global row = b*T + t

  for (int i = tid; i < 64 * DD; i += 256) {
    const int row = i >> 8, col = i & (DD - 1);
    xs[row * XPAD + col] = (__bf16)x[(size_t)(r0 + row) * DD + col];
  }

  const int wave  = tid >> 5;
  const int mtile = wave >> 1;          // 0..3
  const int nbase = (wave & 1) * 8;     // n-tiles nbase..nbase+7

  v8f   acc[8] = {};
  v16bf a[8];

  for (int kp = 0; kp < 2; ++kp) {
    __syncthreads();                    // xs ready / previous us half consumed
    for (int i = tid; i < 128 * DD; i += 256) {
      const int kl = i >> 8, n = i & (DD - 1);
      const int kg = kp * 128 + kl;
      us[(kl >> 5) * (DD * 32) + n * 32 + (kl & 31)] = (__bf16)U[(size_t)kg * DD + n];
    }
    __syncthreads();
    if (kp == 0) {
#pragma unroll
      for (int k = 0; k < 8; ++k) a[k] = load_A_frag(xs, mtile * 16, k * 32, XPAD);
    }
#pragma unroll
    for (int nt = 0; nt < 8; ++nt) {
#pragma unroll
      for (int k = 0; k < 4; ++k) {
        const v16bf bfrag = load_B_frag(us, k, nbase + nt);
        acc[nt] = WMMA_BF16(a[kp * 4 + k], bfrag, acc[nt]);
      }
    }
  }

  const int lane = tid & 31;
  const int l16  = lane & 15;
  const int hi   = (lane >> 4) & 1;
#pragma unroll
  for (int nt = 0; nt < 8; ++nt) {
    const int n  = (nbase + nt) * 16 + l16;
    const float bn = bias[n];
#pragma unroll
    for (int r = 0; r < 8; ++r) {
      const int mrow = mtile * 16 + hi * 8 + r;
      const int rg   = r0 + mrow;            // = b*T + t
      const int bidx = rg >> 10;             // T = 1024
      const int t    = rg & (TT - 1);
      Xu[((size_t)t * BB + bidx) * DD + n] = acc[nt][r] + bn;
    }
  }
}

// =====================================================================
// Kernel 2: serial recurrence. One persistent workgroup (16 wave32 = 1 WGP).
// State bf16 double-buffered in LDS; W in registers as B-fragments.
// Xu slices streamed into LDS by the Tensor Data Mover (double-buffered),
// hidden behind the per-step WMMA chain; s_wait_tensorcnt + barrier publish.
// =====================================================================
__global__ __launch_bounds__(512) void recurrence_kernel(
    const float* __restrict__ s0, const float* __restrict__ W,
    const float* __restrict__ Xu, float* __restrict__ out) {
  __shared__ __bf16 st[2][BB * XPAD];   // 2 x ~34 KB
#if defined(HAVE_TDM)
  __shared__ float xub[2][BB * DD];     // 2 x 64 KB TDM landing buffers
#endif

  const int tid  = threadIdx.x;
  const int wave = tid >> 5;            // 0..15 == n-tile owned by this wave
  const int lane = tid & 31;
  const int l16  = lane & 15;
  const int hi   = (lane >> 4) & 1;
  const int n    = wave * 16 + l16;     // this lane's output column

  // W as per-wave B-fragment registers (8 ktiles x 8 VGPRs, persistent)
  v16bf wf[8];
#pragma unroll
  for (int k = 0; k < 8; ++k) {
#pragma unroll
    for (int j = 0; j < 16; ++j) {
      wf[k][j] = (__bf16)W[(size_t)(k * 32 + hi * 16 + j) * DD + n];
    }
  }

  for (int i = tid; i < BB * DD; i += 512) {
    const int row = i >> 8, col = i & (DD - 1);
    st[0][row * XPAD + col] = (__bf16)s0[i];
  }
#if defined(HAVE_TDM)
  if (wave == 0) tdm_load_slice(Xu, (unsigned)(uintptr_t)&xub[0][0]);
  __builtin_amdgcn_s_wait_tensorcnt(0);
#endif
  __syncthreads();

  float* scan = out + BB * DD;          // out = [final_state | scan_out[B,T,D]]
  int cur = 0;

  for (int t = 0; t < TT; ++t) {
#if defined(HAVE_TDM)
    if (wave == 0 && t + 1 < TT)        // DMA next 64 KB slice behind the WMMA chain
      tdm_load_slice(Xu + (size_t)(t + 1) * BB * DD,
                     (unsigned)(uintptr_t)&xub[(t + 1) & 1][0]);
#else
    const float* xu_t = Xu + (size_t)t * BB * DD;
    if (t + 1 < TT)
      __builtin_prefetch(Xu + (size_t)(t + 1) * BB * DD + tid * 32, 0, 0);
#endif

    v8f acc[4] = {};
#pragma unroll
    for (int m = 0; m < 4; ++m) {
#pragma unroll
      for (int k = 0; k < 8; ++k) {
        const v16bf afrag = load_A_frag(st[cur], m * 16, k * 32, XPAD);
        acc[m] = WMMA_BF16(afrag, wf[k], acc[m]);
      }
    }

    const int nxt = cur ^ 1;
#pragma unroll
    for (int m = 0; m < 4; ++m) {
#pragma unroll
      for (int r = 0; r < 8; ++r) {
        const int row = m * 16 + hi * 8 + r;                 // batch index b
#if defined(HAVE_TDM)
        const float xv = xub[t & 1][row * DD + n];
#else
        const float xv = xu_t[row * DD + n];
#endif
        const float v = FAST_TANH(acc[m][r] + xv);
        st[nxt][row * XPAD + n] = (__bf16)v;                 // next-step state
        scan[((size_t)row * TT + t) * DD + n] = v;           // scan_out[b,t,n]
        if (t == TT - 1) out[row * DD + n] = v;              // final_state
      }
    }
#if defined(HAVE_TDM)
    __builtin_amdgcn_s_wait_tensorcnt(0);   // wave0's DMA landed; others no-op
#endif
    __syncthreads();                        // state + xu buffer handoff
    cur = nxt;
  }
}

extern "C" void kernel_launch(void* const* d_in, const int* in_sizes, int n_in,
                              void* d_out, int out_size, void* d_ws, size_t ws_size,
                              hipStream_t stream) {
  // setup_inputs order: s0 [B,D], x [B,T,D], W [D,D], U [D,D], b [D]
  const float* s0 = (const float*)d_in[0];
  const float* x  = (const float*)d_in[1];
  const float* W  = (const float*)d_in[2];
  const float* U  = (const float*)d_in[3];
  const float* b  = (const float*)d_in[4];
  float* out = (float*)d_out;
  float* Xu  = (float*)d_ws;            // [T, B, D] f32 = 64 MB scratch

  xu_gemm_kernel<<<(BB * TT) / 64, 256, 0, stream>>>(x, U, b, Xu);
  recurrence_kernel<<<1, 512, 0, stream>>>(s0, W, Xu, out);
}
